// CoreAttention_20933670601021
// MI455X (gfx1250) — matrix operand order repaired and checked by reading, not compiled, not run on hardware
//
#include <hip/hip_runtime.h>

// ---------------------------------------------------------------------------
// Causal attention (B=2,H=16,S=2048,D=128, fp32) — CDNA5 WMMA flash attention.
//   Pre-pass (once, into d_ws):  Kh = bf16(K)  [bh][kv][d]
//                                Vt = bf16(V^T) [bh][d][kv]  (LDS-tiled)
//   Main loop per 32-kv group: 16 QK WMMAs (K_hi x (Q_hi+Q_lo), Q split done
//   once per wave => ~fp32 scores) + 8 PV WMMAs (P_bf16 x V_hi). All operand
//   fragments are contiguous b128 loads; no in-loop converts, no gathers.
//   Both score-tile WMMA chains run unconditionally & interleaved (2-way ILP);
//   no LDS/barriers in the attention kernel: K+V fit in the 192MB L2.
// ---------------------------------------------------------------------------

typedef __attribute__((ext_vector_type(16))) __bf16 v16bf;
typedef __attribute__((ext_vector_type(8)))  __bf16 v8bf;
typedef __attribute__((ext_vector_type(4)))  __bf16 v4bf;
typedef __attribute__((ext_vector_type(8)))  float  v8f;
typedef __attribute__((ext_vector_type(4)))  float  f32x4;

constexpr int B_ = 2, H_ = 16, S_ = 2048, D_ = 128;
constexpr float SCALE = 0.08838834764831845f;   // 1/sqrt(128)
constexpr float MASKV = -10000.0f;

#define SHUF16(a, b) __builtin_shufflevector(a, b, 0,1,2,3,4,5,6,7,8,9,10,11,12,13,14,15)
#define V8F_ZERO (v8f){0.f,0.f,0.f,0.f,0.f,0.f,0.f,0.f}

__device__ __forceinline__ v8f wmma_bf16(v16bf a, v16bf b, v8f c) {
  return __builtin_amdgcn_wmma_f32_16x16x32_bf16(false, a, false, b,
                                                 (short)0, c, false, false);
}

// fp32 -> bf16 (hi only), 8 contiguous, into frag elems [e0..e0+7]  (fallback)
__device__ __forceinline__ void cvt8h(const float* __restrict__ p, v16bf& hi, int e0) {
  f32x4 a = *(const f32x4*)(p);
  f32x4 b = *(const f32x4*)(p + 4);
  float x[8] = {a[0], a[1], a[2], a[3], b[0], b[1], b[2], b[3]};
#pragma unroll
  for (int i = 0; i < 8; ++i) hi[e0 + i] = (__bf16)x[i];
}

// fp32 -> bf16 hi + residual lo (used only for Q, once per wave)
__device__ __forceinline__ void cvt8hl(const float* __restrict__ p,
                                       v16bf& hi, v16bf& lo, int e0) {
  f32x4 a = *(const f32x4*)(p);
  f32x4 b = *(const f32x4*)(p + 4);
  float x[8] = {a[0], a[1], a[2], a[3], b[0], b[1], b[2], b[3]};
#pragma unroll
  for (int i = 0; i < 8; ++i) {
    __bf16 h = (__bf16)x[i];
    hi[e0 + i] = h;
    lo[e0 + i] = (__bf16)(x[i] - (float)h);
  }
}

// Two S^T tiles (16 kv x 16 q) for kv-tiles j0 and j0+1, chains interleaved.
// Tile j0+1's rows are always in-bounds (max row = m0+31 <= S-1).
template <bool PRE>
__device__ __forceinline__ void score_pair(const float* __restrict__ kf32,
                                           const __bf16* __restrict__ kbf,
                                           const v16bf (&qh)[4], const v16bf (&ql)[4],
                                           int j0, int ln, int hh, v8f& c0, v8f& c1) {
  c0 = V8F_ZERO;
  c1 = V8F_ZERO;
#pragma unroll
  for (int s = 0; s < 4; ++s) {             // 4 slices of 32 over D=128
    v16bf k0, k1;
    if constexpr (PRE) {
      const __bf16* kp0 = kbf + (size_t)(j0 * 16 + ln) * D_ + s * 32 + hh * 8;
      const __bf16* kp1 = kp0 + 16 * D_;
      v8bf a0 = *(const v8bf*)kp0;          // A-frag: K {0..7 | 8..15}
      v8bf b0 = *(const v8bf*)(kp0 + 16);   //         K {16..23 | 24..31}
      v8bf a1 = *(const v8bf*)kp1;
      v8bf b1 = *(const v8bf*)(kp1 + 16);
      k0 = SHUF16(a0, b0);
      k1 = SHUF16(a1, b1);
    } else {
      const float* kp0 = kf32 + (size_t)(j0 * 16 + ln) * D_ + s * 32 + hh * 8;
      const float* kp1 = kp0 + 16 * D_;
      cvt8h(kp0,      k0, 0);  cvt8h(kp0 + 16, k0, 8);
      cvt8h(kp1,      k1, 0);  cvt8h(kp1 + 16, k1, 8);
    }
    c0 = wmma_bf16(k0, qh[s], c0);          // two independent 8-deep chains,
    c1 = wmma_bf16(k1, qh[s], c1);          // explicitly interleaved
    c0 = wmma_bf16(k0, ql[s], c0);
    c1 = wmma_bf16(k1, ql[s], c1);
  }
}

template <bool PRE>
__global__ void __launch_bounds__(256)
attn_fwd(const float* __restrict__ Q, const float* __restrict__ K,
         const float* __restrict__ V, const __bf16* __restrict__ KH,
         const __bf16* __restrict__ VT, float* __restrict__ O) {
  const int lane = threadIdx.x & 31;
  const int wave = threadIdx.x >> 5;
  const int ln   = lane & 15;               // N-dim lane (q col / d col)
  const int hh   = lane >> 4;               // lane half

  const int bh   = blockIdx.x >> 4;         // [0,32)  batch*head
  const int qblk = 15 - (blockIdx.x & 15);  // heavy q-blocks launch first
  const int m0   = qblk * 128 + wave * 16;  // this wave's 16 q rows

  const size_t base = (size_t)bh * S_ * D_;
  const float*  q   = Q + base;
  const float*  k32 = K + base;
  const float*  v32 = V + base;
  const __bf16* kbf = KH + base;            // [kv][d] bf16
  const __bf16* vt  = VT + base;            // [d][kv] bf16 (row stride S_)
  float*        o   = O + base;

  // ---- Q fragments (B-operand layout: d {8h..8h+7 | 16+8h..16+8h+7} per lane half) ----
  v16bf qh[4], ql[4];
#pragma unroll
  for (int s = 0; s < 4; ++s) {
    const float* qp = q + (size_t)(m0 + ln) * D_ + s * 32 + hh * 8;
    cvt8hl(qp,     qh[s], ql[s], 0);
    cvt8hl(qp + 16, qh[s], ql[s], 8);
  }

  v8f acc[8];
#pragma unroll
  for (int t = 0; t < 8; ++t) acc[t] = V8F_ZERO;
  float rmax = -1.0e30f;
  float rsum = 0.0f;

  const int ntiles  = (m0 >> 4) + 1;        // causal kv tiles of 16
  const int diag    = m0 >> 4;
  const int ngroups = (ntiles + 1) >> 1;    // 32 kv per group

  for (int g = 0; g < ngroups; ++g) {
    const int j0   = 2 * g, j1 = 2 * g + 1;
    const bool inv1 = (j1 >= ntiles);       // wave-uniform, applied as mask

    if constexpr (PRE) {                    // branchless next-group K prefetch
      int pj = j0 + 2;
      if (pj > S_ / 16 - 2) pj = S_ / 16 - 2;
      const char* pf = (const char*)(kbf + (size_t)pj * 16 * D_) + (size_t)lane * 256;
      __builtin_prefetch(pf, 0, 1);
    }

    v8f s0, s1;
    score_pair<PRE>(k32, kbf, qh, ql, j0, ln, hh, s0, s1);

    // ---- V fragments issued early: latency hides under the softmax VALU ----
    const int kb = g * 32 + hh * 8;         // B-frag: kv {8h..8h+7 | 16+8h..16+8h+7}
    v16bf vf[8];
#pragma unroll
    for (int t = 0; t < 8; ++t) {
      if constexpr (PRE) {
        const __bf16* vp = vt + (size_t)(t * 16 + ln) * S_ + kb;
        v8bf a  = *(const v8bf*)vp;
        v8bf b2 = *(const v8bf*)(vp + 16);
        vf[t] = SHUF16(a, b2);
      } else {
        const float* vp = v32 + (size_t)kb * D_ + t * 16 + ln;
#pragma unroll
        for (int e = 0; e < 16; ++e) vf[t][e] = (__bf16)vp[(size_t)(e + (e & 8)) * D_];
      }
    }

    // scale + causal mask (S^T layout: kv = j*16 + r + 8*hh, q = m0 + ln)
    float t0[8], t1[8];
#pragma unroll
    for (int r = 0; r < 8; ++r) {
      const int kvrel = r + 8 * hh;
      float x = s0[r] * SCALE;
      if (j0 == diag && kvrel > ln) x = MASKV;
      t0[r] = x;
      float y = s1[r] * SCALE;
      if ((j1 == diag && kvrel > ln) || inv1) y = MASKV;
      t1[r] = y;
    }

    // online softmax stats for this lane's q column (pair-reduce over halves)
    float gm = t0[0];
#pragma unroll
    for (int r = 0; r < 8; ++r) { gm = fmaxf(gm, t0[r]); gm = fmaxf(gm, t1[r]); }
    gm = fmaxf(gm, __shfl_xor(gm, 16, 32));
    const float nm = fmaxf(rmax, gm);
    const float sc = __expf(rmax - nm);
    rmax = nm;

    float e0[8], e1[8], ls = 0.f;
#pragma unroll
    for (int r = 0; r < 8; ++r) {
      e0[r] = __expf(t0[r] - nm);
      e1[r] = __expf(t1[r] - nm);
      ls += e0[r] + e1[r];
    }
    ls += __shfl_xor(ls, 16, 32);
    rsum = rsum * sc + ls;

    // P as A-frag: purely in-lane thanks to the S^T trick
    v16bf p;
#pragma unroll
    for (int r = 0; r < 8; ++r) { p[r] = (__bf16)e0[r]; p[r + 8] = (__bf16)e1[r]; }

    // rescale O accumulators (scale for q row r+8*hh lives in that lane id)
    v8f qsv;
#pragma unroll
    for (int r = 0; r < 8; ++r) qsv[r] = __shfl(sc, r + 8 * hh, 32);
#pragma unroll
    for (int t = 0; t < 8; ++t) acc[t] = acc[t] * qsv;

    // PV: O += P(16x32) x V(32x16) per 16-d tile — 8 independent WMMAs
#pragma unroll
    for (int t = 0; t < 8; ++t) acc[t] = wmma_bf16(p, vf[t], acc[t]);
  }

  // epilogue: divide by softmax denom, store (row = r + 8*hh, col = t*16+ln)
  v8f isv;
#pragma unroll
  for (int r = 0; r < 8; ++r) isv[r] = 1.0f / __shfl(rsum, r + 8 * hh, 32);
#pragma unroll
  for (int t = 0; t < 8; ++t) {
    v8f res = acc[t] * isv;
#pragma unroll
    for (int r = 0; r < 8; ++r)
      o[(size_t)(m0 + r + 8 * hh) * D_ + t * 16 + ln] = res[r];
  }
}

// ---- pre-pass 1: elementwise fp32 -> bf16 (K, same layout) ----
__global__ void __launch_bounds__(256)
conv_bf16(const float* __restrict__ in, __bf16* __restrict__ out, int n4) {
  int i = blockIdx.x * 256 + threadIdx.x;
  if (i < n4) {
    f32x4 x = ((const f32x4*)in)[i];
    v4bf y;
#pragma unroll
    for (int e = 0; e < 4; ++e) y[e] = (__bf16)x[e];
    ((v4bf*)out)[i] = y;
  }
}

// ---- pre-pass 2: V [bh][kv][d] fp32 -> Vt [bh][d][kv] bf16, LDS-tiled ----
__global__ void __launch_bounds__(256)
transpose_v(const float* __restrict__ V, __bf16* __restrict__ VT) {
  __shared__ unsigned short sh[32][66];     // [d][kv], pad dodges bank conflicts
  const int bh  = blockIdx.z;
  const int kv0 = blockIdx.x * 64;
  const int d0  = blockIdx.y * 32;
  const float* src = V + (size_t)bh * S_ * D_;
#pragma unroll
  for (int i = 0; i < 8; ++i) {             // 64x32 tile, 8 elems/thread, coalesced
    int idx = threadIdx.x + i * 256;
    int d = idx & 31, kv = idx >> 5;
    float x = src[(size_t)(kv0 + kv) * D_ + d0 + d];
    sh[d][kv] = __builtin_bit_cast(unsigned short, (__bf16)x);
  }
  __syncthreads();
  unsigned int* dst = (unsigned int*)(VT + (size_t)bh * D_ * S_);
#pragma unroll
  for (int i = 0; i < 4; ++i) {             // write rows, u32-packed, coalesced
    int idx = threadIdx.x + i * 256;
    int kvp = idx & 31, d = idx >> 5;
    unsigned int w = (unsigned int)sh[d][2 * kvp] |
                     ((unsigned int)sh[d][2 * kvp + 1] << 16);
    dst[(((size_t)(d0 + d) * S_ + kv0) >> 1) + kvp] = w;
  }
}

extern "C" void kernel_launch(void* const* d_in, const int* /*in_sizes*/, int /*n_in*/,
                              void* d_out, int /*out_size*/, void* d_ws, size_t ws_size,
                              hipStream_t stream) {
  const float* Q = (const float*)d_in[0];
  const float* K = (const float*)d_in[1];
  const float* V = (const float*)d_in[2];
  float*       O = (float*)d_out;

  const size_t nEl   = (size_t)B_ * H_ * S_ * D_;   // 8,388,608
  const size_t bytes = nEl * sizeof(short);         // 16 MB per bf16 tensor
  __bf16* KH = (__bf16*)d_ws;
  __bf16* VT = (__bf16*)((char*)d_ws + bytes);

  dim3 grid(B_ * H_ * (S_ / 128));                  // 512 blocks, 8 waves each

  if (ws_size >= 2 * bytes) {
    conv_bf16<<<(unsigned)((nEl / 4 + 255) / 256), 256, 0, stream>>>(K, KH, (int)(nEl / 4));
    transpose_v<<<dim3(S_ / 64, D_ / 32, B_ * H_), 256, 0, stream>>>(V, VT);
    attn_fwd<true><<<grid, 256, 0, stream>>>(Q, K, V, KH, VT, O);
  } else {
    attn_fwd<false><<<grid, 256, 0, stream>>>(Q, K, V, KH, VT, O);
  }
}
